// Bert_CRF_32109175505529
// MI455X (gfx1250) — matrix operand order repaired and checked by reading, not compiled, not run on hardware
//
#include <hip/hip_runtime.h>
#include <hip/hip_bf16.h>
#include <math.h>

#define Hd   768
#define Sd   256
#define Bd   16
#define NHd  12
#define Ld   12
#define FFd  3072
#define Td   9
#define NTOK (Bd * Sd)

typedef __attribute__((ext_vector_type(16))) __bf16 v16bf;
typedef __attribute__((ext_vector_type(8)))  float  v8f;
typedef int v4i __attribute__((vector_size(16)));
typedef __attribute__((address_space(1))) v4i* gv4p;   // global (device) AS
typedef __attribute__((address_space(3))) v4i* lv4p;   // LDS AS

union FragBF { unsigned u[8]; v16bf v; };

#if defined(__has_builtin)
#if __has_builtin(__builtin_amdgcn_global_load_async_to_lds_b128)
#define HAVE_ASYNC_LDS 1
#endif
#if __has_builtin(__builtin_amdgcn_s_wait_asynccnt)
#define HAVE_WAIT_ASYNC 1
#endif
#endif

__device__ __forceinline__ unsigned short f2bf(float f) {
  unsigned u = __builtin_bit_cast(unsigned, f);
  u += 0x7FFFu + ((u >> 16) & 1u);   // round-to-nearest-even
  return (unsigned short)(u >> 16);
}
__device__ __forceinline__ unsigned pack2bf(float a, float b) {
  return (unsigned)f2bf(a) | ((unsigned)f2bf(b) << 16);
}

// ---------------------------------------------------------------------------
// GEMM: out[M,N] = act(A_bf16[M,K] @ W_f32[K,N] + bias)   (bf16 and/or f32 out)
// Block: 256 thr = 8 waves, tile M=64 N=128; wave = 16(M) x 64(N) = 4 WMMA accs
// A tile staged HBM->LDS via async-LDS DMA (ASYNCcnt); W tile converted
// fp32->bf16 on the VALU while the async copy is in flight; next-K W tile
// prefetched into L2 (global_prefetch_b8).
// ---------------------------------------------------------------------------
__global__ __launch_bounds__(256) void gemm_bf16_kernel(
    const unsigned short* __restrict__ A, const float* __restrict__ W,
    const float* __restrict__ bias, unsigned short* __restrict__ outBf,
    float* __restrict__ outF, int M, int N, int K, int act)
{
  __shared__ unsigned Asm[64 * 16];   // 64 rows x 16 bf16-pairs (K-tile 32)
  __shared__ unsigned Bsm[128 * 16];  // 128 cols x 16 bf16-pairs (transposed)
  const int tid  = threadIdx.x;
  const int lane = tid & 31, wave = tid >> 5;
  const int half = lane >> 4, c = lane & 15;
  const int wm = wave >> 1, wn = wave & 1;
  const int blockM = blockIdx.y * 64;
  const int blockN = blockIdx.x * 128;
  const unsigned* A32 = (const unsigned*)A;

  v8f zero = {0.f,0.f,0.f,0.f,0.f,0.f,0.f,0.f};
  v8f acc[4] = {zero, zero, zero, zero};

  for (int k0 = 0; k0 < K; k0 += 32) {
    __syncthreads();
#if defined(HAVE_ASYNC_LDS)
    {   // async DMA: 256 lanes x 16B = whole 64x32 bf16 A tile
      int row = tid >> 2, seg = tid & 3;
      const unsigned* gsrc = &A32[(size_t)(blockM + row) * (K >> 1) + (k0 >> 1) + seg * 4];
      unsigned* ldst = &Asm[row * 16 + seg * 4];
      __builtin_amdgcn_global_load_async_to_lds_b128(
          (gv4p)(unsigned long long)(const void*)gsrc,
          (lv4p)(unsigned)(unsigned long long)(const void*)ldst,
          0, 0);
    }
#else
    for (int i = tid; i < 64 * 16; i += 256) {      // fallback: sync stage A
      int row = i >> 4, kp = i & 15;
      Asm[i] = A32[(size_t)(blockM + row) * (K >> 1) + (k0 >> 1) + kp];
    }
#endif
    for (int i = tid; i < 128 * 16; i += 256) {     // stage W fp32 -> bf16 pairs
      int p = i >> 7, n = i & 127;
      float f0 = W[(size_t)(k0 + 2 * p) * N + blockN + n];
      float f1 = W[(size_t)(k0 + 2 * p + 1) * N + blockN + n];
      Bsm[n * 16 + p] = pack2bf(f0, f1);
    }
    if (k0 + 32 < K) {   // L2 prefetch of next K-step's weight tile
      const float* nw = &W[(size_t)(k0 + 32 + (tid >> 3)) * N + blockN + (tid & 7) * 16];
      __builtin_prefetch(nw, 0, 1);
    }
#if defined(HAVE_ASYNC_LDS)
#if defined(HAVE_WAIT_ASYNC)
    __builtin_amdgcn_s_wait_asynccnt(0);
#else
    asm volatile("s_wait_asynccnt 0" ::: "memory");
#endif
#endif
    __syncthreads();

    FragBF af;
    #pragma unroll
    for (int v = 0; v < 8; ++v) {
      int kp = half * 8 + v;                        // A-fragment K-pair layout
      af.u[v] = Asm[(wm * 16 + c) * 16 + kp];
    }
    #pragma unroll
    for (int j = 0; j < 4; ++j) {
      FragBF bf;
      #pragma unroll
      for (int v = 0; v < 8; ++v) {
        int kp = half * 8 + v;                      // B-fragment K-pair layout
        bf.u[v] = Bsm[(wn * 64 + j * 16 + c) * 16 + kp];
      }
      acc[j] = __builtin_amdgcn_wmma_f32_16x16x32_bf16(
          false, af.v, false, bf.v, (short)0, acc[j], false, false);
    }
  }

  #pragma unroll
  for (int j = 0; j < 4; ++j) {
    #pragma unroll
    for (int r = 0; r < 8; ++r) {
      int mrow = blockM + wm * 16 + r + half * 8;   // C layout: row r+8*half
      int ncol = blockN + wn * 64 + j * 16 + c;
      float val = acc[j][r] + bias[ncol];
      if (act == 1) val = 0.5f * val * (1.0f + erff(val * 0.70710678118654752f));
      size_t idx = (size_t)mrow * N + ncol;
      if (outF)  outF[idx]  = val;
      if (outBf) outBf[idx] = f2bf(val);
    }
  }
}

// ---------------------------------------------------------------------------
// Attention: one block per (batch, head). 8 waves x 16 q-rows, 2 passes = 256.
// Scores in registers, softmax via half-wave shuffles, P relayout via LDS.
// ---------------------------------------------------------------------------
__global__ __launch_bounds__(256) void attn_kernel(
    const unsigned short* __restrict__ qb, const unsigned short* __restrict__ kb,
    const unsigned short* __restrict__ vb, const int* __restrict__ amask,
    unsigned short* __restrict__ ctx)
{
  __shared__ unsigned short vT[64][256];      // V transposed: [d][s]   32 KB
  __shared__ unsigned short pst[8][16][32];   // per-wave P staging      8 KB
  const int b  = blockIdx.x / NHd;
  const int hh = blockIdx.x % NHd;
  const int tid = threadIdx.x;
  const int lane = tid & 31, wave = tid >> 5;
  const int half = lane >> 4, c = lane & 15;

  for (int i = tid; i < 64 * 256; i += 256) {
    int d = i >> 8, s = i & 255;
    vT[d][s] = vb[(size_t)(b * Sd + s) * Hd + hh * 64 + d];
  }
  __syncthreads();

  const unsigned* q32 = (const unsigned*)qb;
  const unsigned* k32 = (const unsigned*)kb;
  const float scale = 0.125f;  // 1/sqrt(64)
  v8f zero = {0.f,0.f,0.f,0.f,0.f,0.f,0.f,0.f};

  for (int it = 0; it < 2; ++it) {
    int m0 = (it * 8 + wave) * 16;
    v8f sacc[16];
    #pragma unroll
    for (int j = 0; j < 16; ++j) sacc[j] = zero;

    FragBF aq[2];
    #pragma unroll
    for (int t = 0; t < 2; ++t)
      #pragma unroll
      for (int v = 0; v < 8; ++v) {
        int kp = t * 16 + half * 8 + v;
        aq[t].u[v] = q32[((size_t)(b * Sd + m0 + c) * Hd + hh * 64) / 2 + kp];
      }
    for (int j = 0; j < 16; ++j) {      // S = Q K^T over d (K=64, 2 steps)
      #pragma unroll
      for (int t = 0; t < 2; ++t) {
        FragBF bk;
        #pragma unroll
        for (int v = 0; v < 8; ++v) {
          int kp = t * 16 + half * 8 + v;
          bk.u[v] = k32[((size_t)(b * Sd + j * 16 + c) * Hd + hh * 64) / 2 + kp];
        }
        sacc[j] = __builtin_amdgcn_wmma_f32_16x16x32_bf16(
            false, aq[t].v, false, bk.v, (short)0, sacc[j], false, false);
      }
    }
    #pragma unroll
    for (int j = 0; j < 16; ++j) {
      float mbias = (amask[b * Sd + j * 16 + c] > 0) ? 0.f : -1.0e9f;
      #pragma unroll
      for (int r = 0; r < 8; ++r) sacc[j][r] = sacc[j][r] * scale + mbias;
    }
    // softmax over 256 cols per row (row r+8*half, cols across half-wave lanes)
    #pragma unroll
    for (int r = 0; r < 8; ++r) {
      float mx = -3.4e38f;
      #pragma unroll
      for (int j = 0; j < 16; ++j) mx = fmaxf(mx, sacc[j][r]);
      #pragma unroll
      for (int m = 8; m >= 1; m >>= 1) mx = fmaxf(mx, __shfl_xor(mx, m, 32));
      float sm = 0.f;
      #pragma unroll
      for (int j = 0; j < 16; ++j) { float p = __expf(sacc[j][r] - mx); sacc[j][r] = p; sm += p; }
      #pragma unroll
      for (int m = 8; m >= 1; m >>= 1) sm += __shfl_xor(sm, m, 32);
      float inv = 1.f / sm;
      #pragma unroll
      for (int j = 0; j < 16; ++j) sacc[j][r] *= inv;
    }
    // ctx = P @ V  (K = 256, 8 steps; P relayout C->A via per-wave LDS tile)
    v8f cacc[4] = {zero, zero, zero, zero};
    for (int g = 0; g < 8; ++g) {
      #pragma unroll
      for (int jj = 0; jj < 2; ++jj) {
        int j = 2 * g + jj;
        #pragma unroll
        for (int r = 0; r < 8; ++r)
          pst[wave][r + half * 8][jj * 16 + c] = f2bf(sacc[j][r]);
      }
      asm volatile("s_wait_dscnt 0" ::: "memory");
      FragBF ap;
      #pragma unroll
      for (int v = 0; v < 8; ++v) {
        int kp = half * 8 + v;
        ap.u[v] = ((const unsigned*)&pst[wave][0][0])[c * 16 + kp];
      }
      #pragma unroll
      for (int jd = 0; jd < 4; ++jd) {
        FragBF bv;
        #pragma unroll
        for (int v = 0; v < 8; ++v) {
          int kp = g * 16 + half * 8 + v;
          bv.u[v] = ((const unsigned*)&vT[jd * 16 + c][0])[kp];
        }
        cacc[jd] = __builtin_amdgcn_wmma_f32_16x16x32_bf16(
            false, ap.v, false, bv.v, (short)0, cacc[jd], false, false);
      }
    }
    #pragma unroll
    for (int jd = 0; jd < 4; ++jd)
      #pragma unroll
      for (int r = 0; r < 8; ++r) {
        int srow = m0 + r + half * 8;
        ctx[(size_t)(b * Sd + srow) * Hd + hh * 64 + jd * 16 + c] = f2bf(cacc[jd][r]);
      }
  }
}

// ---------------------------------------------------------------------------
// LayerNorm helpers / kernels (1 block per token, 256 threads, 3 elems each)
// ---------------------------------------------------------------------------
__device__ __forceinline__ float block_sum256(float v, float* sred) {
  int tid = threadIdx.x;
  #pragma unroll
  for (int m = 16; m >= 1; m >>= 1) v += __shfl_xor(v, m, 32);
  if ((tid & 31) == 0) sred[tid >> 5] = v;
  __syncthreads();
  float tot = 0.f;
  #pragma unroll
  for (int w = 0; w < 8; ++w) tot += sred[w];
  __syncthreads();
  return tot;
}

__global__ __launch_bounds__(256) void embed_ln_kernel(
    const int* __restrict__ ids, const int* __restrict__ tts,
    const float* __restrict__ wemb, const float* __restrict__ pemb,
    const float* __restrict__ temb, const float* __restrict__ gg,
    const float* __restrict__ bb, float* __restrict__ hF,
    unsigned short* __restrict__ hB)
{
  __shared__ float sred[8];
  int tok = blockIdx.x, tid = threadIdx.x;
  int s = tok & 255;
  int id = ids[tok], ty = tts[tok];
  float x[3];
  #pragma unroll
  for (int i = 0; i < 3; ++i) {
    int h = tid + i * 256;
    x[i] = wemb[(size_t)id * Hd + h] + pemb[(size_t)s * Hd + h] + temb[(size_t)ty * Hd + h];
  }
  float mean = block_sum256(x[0] + x[1] + x[2], sred) * (1.f / Hd);
  float vs = 0.f;
  #pragma unroll
  for (int i = 0; i < 3; ++i) { float d = x[i] - mean; vs += d * d; }
  float var = block_sum256(vs, sred) * (1.f / Hd);
  float inv = rsqrtf(var + 1e-12f);
  #pragma unroll
  for (int i = 0; i < 3; ++i) {
    int h = tid + i * 256;
    float y = (x[i] - mean) * inv * gg[h] + bb[h];
    size_t idx = (size_t)tok * Hd + h;
    hF[idx] = y; hB[idx] = f2bf(y);
  }
}

__global__ __launch_bounds__(256) void add_ln_kernel(
    float* __restrict__ hF, unsigned short* __restrict__ hB,
    const float* __restrict__ tmp, const float* __restrict__ gg,
    const float* __restrict__ bb)
{
  __shared__ float sred[8];
  int tok = blockIdx.x, tid = threadIdx.x;
  float x[3];
  #pragma unroll
  for (int i = 0; i < 3; ++i) {
    size_t idx = (size_t)tok * Hd + tid + i * 256;
    x[i] = hF[idx] + tmp[idx];
  }
  float mean = block_sum256(x[0] + x[1] + x[2], sred) * (1.f / Hd);
  float vs = 0.f;
  #pragma unroll
  for (int i = 0; i < 3; ++i) { float d = x[i] - mean; vs += d * d; }
  float var = block_sum256(vs, sred) * (1.f / Hd);
  float inv = rsqrtf(var + 1e-12f);
  #pragma unroll
  for (int i = 0; i < 3; ++i) {
    int h = tid + i * 256;
    float y = (x[i] - mean) * inv * gg[h] + bb[h];
    size_t idx = (size_t)tok * Hd + h;
    hF[idx] = y; hB[idx] = f2bf(y);
  }
}

// ---------------------------------------------------------------------------
// Dense head: emis[tok][tag] = h . W[:,tag] + b   (tiny, plain VALU)
// ---------------------------------------------------------------------------
__global__ void dense_kernel(const float* __restrict__ hF,
                             const float* __restrict__ W,
                             const float* __restrict__ bias,
                             float* __restrict__ emis)
{
  int idx = blockIdx.x * 256 + threadIdx.x;
  if (idx >= NTOK * Td) return;
  int tok = idx / Td, tg = idx % Td;
  float acc = bias[tg];
  for (int h = 0; h < Hd; ++h) acc += hF[(size_t)tok * Hd + h] * W[h * Td + tg];
  emis[idx] = acc;
}

// ---------------------------------------------------------------------------
// CRF: forward (logZ), numerator score, Viterbi + backtrack. Single block.
// ---------------------------------------------------------------------------
__global__ __launch_bounds__(256) void crf_kernel(
    const float* __restrict__ emis, const int* __restrict__ amask,
    const int* __restrict__ tags, const float* __restrict__ cstart,
    const float* __restrict__ cend, const float* __restrict__ ctrans,
    float* __restrict__ out, int* __restrict__ hist)
{
  __shared__ float alpha[Bd][Td], nxt[Bd][Td], vit[Bd][Td], nvit[Bd][Td];
  __shared__ float trn[Td][Td], st[Td], en[Td];
  __shared__ float scoreB[Bd], logZB[Bd];
  __shared__ int lastT[Bd];
  int tid = threadIdx.x;
  if (tid < Td * Td) trn[tid / Td][tid % Td] = ctrans[tid];
  if (tid < Td) { st[tid] = cstart[tid]; en[tid] = cend[tid]; }
  __syncthreads();
  int b = tid / Td, t = tid % Td;
  bool act = tid < Bd * Td;
  if (act) {
    float a0 = st[t] + emis[(size_t)(b * Sd) * Td + t];
    alpha[b][t] = a0; vit[b][t] = a0;
  }
  __syncthreads();
  for (int s = 1; s < Sd; ++s) {
    if (act) {
      bool mm = amask[b * Sd + s] > 0;
      float e = emis[(size_t)(b * Sd + s) * Td + t];
      float mx = -3.4e38f;
      #pragma unroll
      for (int i = 0; i < Td; ++i) mx = fmaxf(mx, alpha[b][i] + trn[i][t]);
      float sm = 0.f;
      #pragma unroll
      for (int i = 0; i < Td; ++i) sm += __expf(alpha[b][i] + trn[i][t] - mx);
      nxt[b][t] = mm ? (mx + __logf(sm) + e) : alpha[b][t];
      float bm = -3.4e38f; int bi = 0;
      #pragma unroll
      for (int i = 0; i < Td; ++i) {
        float sc = vit[b][i] + trn[i][t];
        if (sc > bm) { bm = sc; bi = i; }
      }
      nvit[b][t] = mm ? (bm + e) : vit[b][t];
      hist[(s - 1) * (Bd * Td) + b * Td + t] = mm ? bi : t;
    }
    __syncthreads();
    if (act) { alpha[b][t] = nxt[b][t]; vit[b][t] = nvit[b][t]; }
    __syncthreads();
  }
  if (tid < Bd) {
    int bb = tid;
    int prev = tags[bb * Sd];
    float sc = st[prev] + emis[(size_t)(bb * Sd) * Td + prev];
    int length = 0;
    for (int s = 0; s < Sd; ++s) length += (amask[bb * Sd + s] > 0) ? 1 : 0;
    for (int s = 1; s < Sd; ++s) {
      int tg = tags[bb * Sd + s];
      if (amask[bb * Sd + s] > 0)
        sc += trn[prev][tg] + emis[(size_t)(bb * Sd + s) * Td + tg];
      prev = tg;
    }
    if (length < 1) length = 1;
    sc += en[tags[bb * Sd + length - 1]];
    scoreB[bb] = sc;
    float mx = -3.4e38f;
    for (int i = 0; i < Td; ++i) mx = fmaxf(mx, alpha[bb][i] + en[i]);
    float sm = 0.f;
    for (int i = 0; i < Td; ++i) sm += __expf(alpha[bb][i] + en[i] - mx);
    logZB[bb] = mx + __logf(sm);
    float bm = -3.4e38f; int bi = 0;
    for (int i = 0; i < Td; ++i) {
      float v = vit[bb][i] + en[i];
      if (v > bm) { bm = v; bi = i; }
    }
    lastT[bb] = bi;
  }
  __syncthreads();
  if (tid == 0) {
    float accl = 0.f;
    for (int i = 0; i < Bd; ++i) accl += scoreB[i] - logZB[i];
    out[0] = -accl / (float)Bd;
  }
  if (tid < Bd) {
    int bb = tid, cur = lastT[bb];
    out[1 + bb * Sd + (Sd - 1)] = (float)cur;
    for (int s = Sd - 2; s >= 0; --s) {
      cur = hist[s * (Bd * Td) + bb * Td + cur];
      out[1 + bb * Sd + s] = (float)cur;
    }
  }
}

// ---------------------------------------------------------------------------
extern "C" void kernel_launch(void* const* d_in, const int* in_sizes, int n_in,
                              void* d_out, int out_size, void* d_ws, size_t ws_size,
                              hipStream_t stream)
{
  (void)in_sizes; (void)n_in; (void)out_size; (void)ws_size;
  const int*   ids    = (const int*)d_in[0];
  const int*   tts    = (const int*)d_in[1];
  const int*   amask  = (const int*)d_in[2];
  const int*   tags   = (const int*)d_in[3];
  const float* wemb   = (const float*)d_in[4];
  const float* pemb   = (const float*)d_in[5];
  const float* temb   = (const float*)d_in[6];
  const float* egln   = (const float*)d_in[7];
  const float* eglb   = (const float*)d_in[8];
  const float* Wq     = (const float*)d_in[9];
  const float* bq     = (const float*)d_in[10];
  const float* Wk     = (const float*)d_in[11];
  const float* bk     = (const float*)d_in[12];
  const float* Wv     = (const float*)d_in[13];
  const float* bv     = (const float*)d_in[14];
  const float* Wo     = (const float*)d_in[15];
  const float* bo     = (const float*)d_in[16];
  const float* ln1g   = (const float*)d_in[17];
  const float* ln1b   = (const float*)d_in[18];
  const float* W1     = (const float*)d_in[19];
  const float* b1     = (const float*)d_in[20];
  const float* W2     = (const float*)d_in[21];
  const float* b2     = (const float*)d_in[22];
  const float* ln2g   = (const float*)d_in[23];
  const float* ln2b   = (const float*)d_in[24];
  const float* denseW = (const float*)d_in[25];
  const float* denseb = (const float*)d_in[26];
  const float* cstart = (const float*)d_in[27];
  const float* cend   = (const float*)d_in[28];
  const float* ctrans = (const float*)d_in[29];

  char* base = (char*)d_ws;
  size_t off = 0;
  auto take = [&](size_t bytes) -> void* {
    off = (off + 255) & ~(size_t)255;
    void* p = base + off; off += bytes; return p;
  };
  float*          hF   = (float*)take((size_t)NTOK * Hd * 4);
  float*          tmpF = (float*)take((size_t)NTOK * Hd * 4);
  unsigned short* hB   = (unsigned short*)take((size_t)NTOK * Hd * 2);
  unsigned short* qB   = (unsigned short*)take((size_t)NTOK * Hd * 2);
  unsigned short* kB   = (unsigned short*)take((size_t)NTOK * Hd * 2);
  unsigned short* vB   = (unsigned short*)take((size_t)NTOK * Hd * 2);
  unsigned short* ctxB = (unsigned short*)take((size_t)NTOK * Hd * 2);
  unsigned short* ffB  = (unsigned short*)take((size_t)NTOK * FFd * 2);
  float*          emis = (float*)take((size_t)NTOK * Td * 4);
  int*            hist = (int*)take((size_t)(Sd - 1) * Bd * Td * 4);

  embed_ln_kernel<<<NTOK, 256, 0, stream>>>(ids, tts, wemb, pemb, temb, egln, eglb, hF, hB);

  dim3 gH(Hd / 128, NTOK / 64);
  dim3 gF(FFd / 128, NTOK / 64);
  for (int l = 0; l < Ld; ++l) {
    const float* wq = Wq + (size_t)l * Hd * Hd;
    const float* wk = Wk + (size_t)l * Hd * Hd;
    const float* wv = Wv + (size_t)l * Hd * Hd;
    const float* wo = Wo + (size_t)l * Hd * Hd;
    const float* w1 = W1 + (size_t)l * Hd * FFd;
    const float* w2 = W2 + (size_t)l * FFd * Hd;
    gemm_bf16_kernel<<<gH, 256, 0, stream>>>(hB, wq, bq + l * Hd, qB, nullptr, NTOK, Hd, Hd, 0);
    gemm_bf16_kernel<<<gH, 256, 0, stream>>>(hB, wk, bk + l * Hd, kB, nullptr, NTOK, Hd, Hd, 0);
    gemm_bf16_kernel<<<gH, 256, 0, stream>>>(hB, wv, bv + l * Hd, vB, nullptr, NTOK, Hd, Hd, 0);
    attn_kernel<<<Bd * NHd, 256, 0, stream>>>(qB, kB, vB, amask, ctxB);
    gemm_bf16_kernel<<<gH, 256, 0, stream>>>(ctxB, wo, bo + l * Hd, nullptr, tmpF, NTOK, Hd, Hd, 0);
    add_ln_kernel<<<NTOK, 256, 0, stream>>>(hF, hB, tmpF, ln1g + l * Hd, ln1b + l * Hd);
    gemm_bf16_kernel<<<gF, 256, 0, stream>>>(hB, w1, b1 + l * FFd, ffB, nullptr, NTOK, FFd, Hd, 1);
    gemm_bf16_kernel<<<gH, 256, 0, stream>>>(ffB, w2, b2 + l * Hd, nullptr, tmpF, NTOK, Hd, FFd, 0);
    add_ln_kernel<<<NTOK, 256, 0, stream>>>(hF, hB, tmpF, ln2g + l * Hd, ln2b + l * Hd);
  }

  dense_kernel<<<(NTOK * Td + 255) / 256, 256, 0, stream>>>(hF, denseW, denseb, emis);
  crf_kernel<<<1, 256, 0, stream>>>(emis, amask, tags, cstart, cend, ctrans,
                                    (float*)d_out, hist);
}
